// LinearAttention_7198365188318
// MI455X (gfx1250) — compile-verified, matrix operand order checked by reading, not run on hardware
//
#include <hip/hip_runtime.h>

// ---------------------------------------------------------------------------
// Types for CDNA5 WMMA (wave32): v_wmma_f32_16x16x32_bf16
// ---------------------------------------------------------------------------
typedef __attribute__((ext_vector_type(16))) __bf16 v16bf;
typedef __attribute__((ext_vector_type(8)))  float  v8f;

union FragBF {
    v16bf v;
    uint4 q[2];   // two 16-byte halves (8 bf16 each)
};

__device__ __forceinline__ v8f vzero8() {
    v8f z = {0.f, 0.f, 0.f, 0.f, 0.f, 0.f, 0.f, 0.f};
    return z;
}

// ---------------------------------------------------------------------------
// CDNA5 async global->LDS copy (ASYNCcnt-tracked). The generic pointer to a
// __shared__ object carries the LDS byte offset in its low 32 bits
// (AS3 -> flat addrspacecast = {aperture_hi, lds_offset}).
// ---------------------------------------------------------------------------
__device__ __forceinline__ void async_copy_b128(void* lds_ptr, const void* gptr) {
    const unsigned lds_off = (unsigned)(uintptr_t)lds_ptr;
    asm volatile("global_load_async_to_lds_b128 %0, %1, off"
                 :: "v"(lds_off), "v"(gptr)
                 : "memory");
}
__device__ __forceinline__ void wait_async0() {
    asm volatile("s_wait_asynccnt 0x0" ::: "memory");
}

// Problem constants
static constexpr int Bb   = 4;
static constexpr int Nn   = 8192;
static constexpr int DIM  = 1024;
static constexpr int H    = 16;
static constexpr int DH   = 64;
static constexpr int QKVC = 3072;          // 3 * H * DH
static constexpr int Mtok = Bb * Nn;       // 32768 tokens
static constexpr float EPS = 1e-12f;
static constexpr float ATTEND_SCALE = 8.0f;

// ---------------------------------------------------------------------------
// 1) RMSNorm: xn = x / max(||x||,eps) * sqrt(dim) * gamma  -> bf16
// ---------------------------------------------------------------------------
__global__ __launch_bounds__(256) void rmsnorm_bf16(const float* __restrict__ x,
                                                    const float* __restrict__ gamma,
                                                    __bf16* __restrict__ xn) {
    __shared__ float red[256];
    const int token = blockIdx.x;
    const int t = threadIdx.x;
    const float4 xv = reinterpret_cast<const float4*>(x + (size_t)token * DIM)[t];
    red[t] = xv.x * xv.x + xv.y * xv.y + xv.z * xv.z + xv.w * xv.w;
    __syncthreads();
    for (int o = 128; o > 0; o >>= 1) {
        if (t < o) red[t] += red[t + o];
        __syncthreads();
    }
    const float inv = 32.0f / fmaxf(sqrtf(red[0]), EPS);   // sqrt(1024) = 32
    const float4 g = reinterpret_cast<const float4*>(gamma)[t];
    __bf16* o = xn + (size_t)token * DIM + t * 4;
    o[0] = (__bf16)(xv.x * inv * g.x);
    o[1] = (__bf16)(xv.y * inv * g.y);
    o[2] = (__bf16)(xv.z * inv * g.z);
    o[3] = (__bf16)(xv.w * inv * g.w);
}

// ---------------------------------------------------------------------------
// 2) fp32 -> bf16 convert (n multiple of 4)
// ---------------------------------------------------------------------------
__global__ __launch_bounds__(256) void f32_to_bf16(const float* __restrict__ src,
                                                   __bf16* __restrict__ dst, int n) {
    const int i = (blockIdx.x * 256 + threadIdx.x) * 4;
    if (i < n) {
        const float4 v = *reinterpret_cast<const float4*>(src + i);
        dst[i + 0] = (__bf16)v.x;
        dst[i + 1] = (__bf16)v.y;
        dst[i + 2] = (__bf16)v.z;
        dst[i + 3] = (__bf16)v.w;
    }
}

// ---------------------------------------------------------------------------
// 3) bf16 GEMM: C[M,N] (f32) = A[M,K] @ B[K,N], row-major.
//    Block tile 128x128, 256 threads = 8 waves, wave tile 32x64 (2x4 WMMA).
//    A staged to LDS with global_load_async_to_lds_b128 (ASYNCcnt path);
//    B staged transposed [N][K] via b128 load + ds scatter; next B k-slice
//    prefetched with global_prefetch.
// ---------------------------------------------------------------------------
__global__ __launch_bounds__(256) void gemm_bf16(const __bf16* __restrict__ A, int lda,
                                                 const __bf16* __restrict__ Bm, int ldb,
                                                 float* __restrict__ C, int ldc, int K) {
    __shared__ __bf16 sA[128][40];
    __shared__ __bf16 sB[128][40];     // [N][K]
    const int tid = threadIdx.x;
    const int m0 = blockIdx.y * 128;
    const int n0 = blockIdx.x * 128;
    const int wave = tid >> 5, lane = tid & 31;
    const int wm = (wave >> 1) * 32;   // 4 wave rows of 32
    const int wn = (wave & 1) * 64;    // 2 wave cols of 64
    const int l15 = lane & 15, half = lane >> 4;

    v8f acc[2][4];
    #pragma unroll
    for (int i = 0; i < 2; ++i)
        #pragma unroll
        for (int j = 0; j < 4; ++j) acc[i][j] = vzero8();

    const int aRow = tid >> 1;
    const int aHalf = (tid & 1) * 16;
    const int bK = tid >> 3;            // 0..31
    const int bN0 = (tid & 7) * 16;     // 0..112

    for (int kc = 0; kc < K; kc += 32) {
        // --- Stage A: 128x32 bf16 via async global->LDS (two b128 per thread)
        const __bf16* ag = A + (size_t)(m0 + aRow) * lda + kc + aHalf;
        async_copy_b128(&sA[aRow][aHalf], ag);
        async_copy_b128(&sA[aRow][aHalf + 8], ag + 8);
        // --- Stage B transposed: 16 contiguous bf16 per thread, scatter to [N][K]
        const __bf16* bg = Bm + (size_t)(kc + bK) * ldb + n0 + bN0;
        uint4 bq0 = reinterpret_cast<const uint4*>(bg)[0];
        uint4 bq1 = reinterpret_cast<const uint4*>(bg)[1];
        const __bf16* bh0 = reinterpret_cast<const __bf16*>(&bq0);
        const __bf16* bh1 = reinterpret_cast<const __bf16*>(&bq1);
        #pragma unroll
        for (int i = 0; i < 8; ++i) sB[bN0 + i][bK] = bh0[i];
        #pragma unroll
        for (int i = 0; i < 8; ++i) sB[bN0 + 8 + i][bK] = bh1[i];
        if (kc + 32 < K)
            __builtin_prefetch(Bm + (size_t)(kc + 32 + bK) * ldb + n0 + bN0, 0, 0);
        wait_async0();
        __syncthreads();

        FragBF a[2], bf[4];
        #pragma unroll
        for (int ti = 0; ti < 2; ++ti) {
            const int row = wm + ti * 16 + l15;
            a[ti].q[0] = *reinterpret_cast<const uint4*>(&sA[row][8 * half]);
            a[ti].q[1] = *reinterpret_cast<const uint4*>(&sA[row][16 + 8 * half]);
        }
        #pragma unroll
        for (int tj = 0; tj < 4; ++tj) {
            const int col = wn + tj * 16 + l15;
            bf[tj].q[0] = *reinterpret_cast<const uint4*>(&sB[col][8 * half]);
            bf[tj].q[1] = *reinterpret_cast<const uint4*>(&sB[col][16 + 8 * half]);
        }
        #pragma unroll
        for (int ti = 0; ti < 2; ++ti)
            #pragma unroll
            for (int tj = 0; tj < 4; ++tj)
                acc[ti][tj] = __builtin_amdgcn_wmma_f32_16x16x32_bf16(
                    false, a[ti].v, false, bf[tj].v, (short)0, acc[ti][tj], false, false);
        __syncthreads();
    }

    // C layout: lane l15 -> column; VGPR j -> row j + 8*half within 16x16 tile
    #pragma unroll
    for (int ti = 0; ti < 2; ++ti)
        #pragma unroll
        for (int tj = 0; tj < 4; ++tj)
            #pragma unroll
            for (int j = 0; j < 8; ++j) {
                const int row = m0 + wm + ti * 16 + j + 8 * half;
                const int col = n0 + wn + tj * 16 + l15;
                C[(size_t)row * ldc + col] = acc[ti][tj][j];
            }
}

// ---------------------------------------------------------------------------
// 4) channel sum-of-squares for q,k over n (split over n with atomics)
// ---------------------------------------------------------------------------
__global__ __launch_bounds__(256) void zero_f32(float* p, int n) {
    const int i = blockIdx.x * 256 + threadIdx.x;
    if (i < n) p[i] = 0.f;
}

__global__ __launch_bounds__(256) void colnorm_partial(const float* __restrict__ qkv,
                                                       float* __restrict__ sumq,
                                                       float* __restrict__ sumk) {
    const int c = blockIdx.x * 256 + threadIdx.x;   // 0..1023
    const int b = blockIdx.y;
    const int n0 = blockIdx.z * 1024;
    const float* base = qkv + (size_t)(b * Nn + n0) * QKVC;
    float sq = 0.f, sk = 0.f;
    for (int n = 0; n < 1024; ++n) {
        const float q = base[(size_t)n * QKVC + c];
        const float k = base[(size_t)n * QKVC + 1024 + c];
        sq += q * q;
        sk += k * k;
    }
    atomicAdd(&sumq[b * 1024 + c], sq);
    atomicAdd(&sumk[b * 1024 + c], sk);
}

__global__ __launch_bounds__(256) void finalize_inv(const float* __restrict__ sumq,
                                                    const float* __restrict__ sumk,
                                                    float* __restrict__ invq,
                                                    float* __restrict__ invk) {
    const int i = blockIdx.x * 256 + threadIdx.x;   // 0..4095
    invq[i] = 1.0f / fmaxf(sqrtf(sumq[i]), EPS);
    invk[i] = 1.0f / fmaxf(sqrtf(sumk[i]), EPS);
}

// ---------------------------------------------------------------------------
// 5) orig_v output: vout[b, h*64+d, n] = qkv[b, n, 2048 + h*64 + d]
// ---------------------------------------------------------------------------
__global__ __launch_bounds__(256) void v_transpose(const float* __restrict__ qkv,
                                                   float* __restrict__ vout) {
    __shared__ float tile[64][65];
    const int n0 = blockIdx.x * 64;
    const int c0 = blockIdx.y * 64;
    const int b = blockIdx.z;
    const int t = threadIdx.x;
    {
        const int cc = t & 63, ng = t >> 6;
        #pragma unroll
        for (int i = 0; i < 16; ++i) {
            const int nn = ng + i * 4;
            tile[nn][cc] = qkv[(size_t)(b * Nn + n0 + nn) * QKVC + 2048 + c0 + cc];
        }
    }
    __syncthreads();
    {
        const int nn = t & 63, cg = t >> 6;
        #pragma unroll
        for (int i = 0; i < 16; ++i) {
            const int cc = cg + i * 4;
            vout[(size_t)(b * (H * DH) + c0 + cc) * Nn + n0 + nn] = tile[nn][cc];
        }
    }
}

// ---------------------------------------------------------------------------
// 6) sim + softmax per (b,h): sim[d,e] = 8 * sum_n q_hat[d,n] k_hat[e,n]
// ---------------------------------------------------------------------------
__global__ __launch_bounds__(256) void sim_softmax(const float* __restrict__ qkv,
                                                   const float* __restrict__ invq,
                                                   const float* __restrict__ invk,
                                                   const float* __restrict__ temp,
                                                   __bf16* __restrict__ attn) {
    __shared__ __bf16 sQ[64][40];
    __shared__ __bf16 sK[64][40];
    __shared__ float sSim[64][65];
    const int h = blockIdx.x;
    const int b = blockIdx.y;
    const int t = threadIdx.x;
    const int wave = t >> 5, lane = t & 31, l15 = lane & 15, half = lane >> 4;
    const int wm = (wave >> 1) * 16;     // 16-row M tile (d)
    const int wn = (wave & 1) * 32;      // two 16-col N tiles (e)

    v8f acc0 = vzero8(), acc1 = vzero8();

    const int d = t & 63;
    const int grp = t >> 6;
    const float tscale = __expf(temp[h]);
    const float qs = invq[b * 1024 + h * DH + d] * tscale;
    const float ks = invk[b * 1024 + h * DH + d];
    const float* qb = qkv + (size_t)b * Nn * QKVC + h * DH;

    for (int n0 = 0; n0 < Nn; n0 += 32) {
        #pragma unroll
        for (int i = 0; i < 8; ++i) {
            const int nn = grp * 8 + i;
            const size_t roff = (size_t)(n0 + nn) * QKVC;
            sQ[d][nn] = (__bf16)(qb[roff + d] * qs);
            sK[d][nn] = (__bf16)(qb[roff + 1024 + d] * ks);
        }
        __syncthreads();
        FragBF a, b0, b1;
        const int ar = wm + l15;
        a.q[0] = *reinterpret_cast<const uint4*>(&sQ[ar][8 * half]);
        a.q[1] = *reinterpret_cast<const uint4*>(&sQ[ar][16 + 8 * half]);
        const int br0 = wn + l15;
        const int br1 = wn + 16 + l15;
        b0.q[0] = *reinterpret_cast<const uint4*>(&sK[br0][8 * half]);
        b0.q[1] = *reinterpret_cast<const uint4*>(&sK[br0][16 + 8 * half]);
        b1.q[0] = *reinterpret_cast<const uint4*>(&sK[br1][8 * half]);
        b1.q[1] = *reinterpret_cast<const uint4*>(&sK[br1][16 + 8 * half]);
        acc0 = __builtin_amdgcn_wmma_f32_16x16x32_bf16(false, a.v, false, b0.v, (short)0, acc0, false, false);
        acc1 = __builtin_amdgcn_wmma_f32_16x16x32_bf16(false, a.v, false, b1.v, (short)0, acc1, false, false);
        __syncthreads();
    }

    #pragma unroll
    for (int j = 0; j < 8; ++j) {
        sSim[wm + j + 8 * half][wn + l15]      = acc0[j];
        sSim[wm + j + 8 * half][wn + 16 + l15] = acc1[j];
    }
    __syncthreads();

    if (t < 64) {
        float mx = -3.4e38f;
        for (int e = 0; e < 64; ++e) mx = fmaxf(mx, sSim[t][e] * ATTEND_SCALE);
        float sum = 0.f;
        for (int e = 0; e < 64; ++e) {
            const float ex = __expf(sSim[t][e] * ATTEND_SCALE - mx);
            sSim[t][e] = ex;
            sum += ex;
        }
        const float r = 1.0f / sum;
        __bf16* arow = attn + ((size_t)((b * H + h) * 64 + t)) * 64;
        for (int e = 0; e < 64; ++e) arow[e] = (__bf16)(sSim[t][e] * r);
    }
}

// ---------------------------------------------------------------------------
// 7) Y[b, n, h*64+d] = sum_e attn[b,h,d,e] * v[b,n, h*64+e]
//    V(128x64) @ attn^T(64x64) via WMMA; attn staged with async copies.
// ---------------------------------------------------------------------------
__global__ __launch_bounds__(256) void apply_attn(const float* __restrict__ qkv,
                                                  const __bf16* __restrict__ attn,
                                                  __bf16* __restrict__ Y) {
    __shared__ __bf16 sV[128][72];
    __shared__ __bf16 sB[64][72];
    const int n0 = blockIdx.x * 128;
    const int h = blockIdx.y;
    const int b = blockIdx.z;
    const int t = threadIdx.x;
    const int wave = t >> 5, lane = t & 31, l15 = lane & 15, half = lane >> 4;
    const int wm = (wave >> 1) * 32;
    const int wn = (wave & 1) * 32;

    {   // stage attn row-major (sB[d][e]) via async global->LDS: 16 bf16/thread
        const __bf16* ab = attn + (size_t)(b * H + h) * 4096;
        const int dd = t >> 2, cc = (t & 3) * 16;   // rows of 64, 16-elem runs
        async_copy_b128(&sB[dd][cc],     ab + dd * 64 + cc);
        async_copy_b128(&sB[dd][cc + 8], ab + dd * 64 + cc + 8);
        // stage V (fp32 -> bf16), coalesced over e
        const int e = t & 63, mg = t >> 6;
        const float* vb = qkv + (size_t)(b * Nn + n0) * QKVC + 2048 + h * DH;
        #pragma unroll
        for (int i = 0; i < 32; ++i) {
            const int m = mg + i * 4;
            sV[m][e] = (__bf16)(vb[(size_t)m * QKVC + e]);
        }
    }
    wait_async0();
    __syncthreads();

    v8f acc[2][2];
    acc[0][0] = vzero8(); acc[0][1] = vzero8();
    acc[1][0] = vzero8(); acc[1][1] = vzero8();

    #pragma unroll
    for (int kc = 0; kc < 64; kc += 32) {
        FragBF a[2], bf[2];
        #pragma unroll
        for (int ti = 0; ti < 2; ++ti) {
            const int row = wm + ti * 16 + l15;
            a[ti].q[0] = *reinterpret_cast<const uint4*>(&sV[row][kc + 8 * half]);
            a[ti].q[1] = *reinterpret_cast<const uint4*>(&sV[row][kc + 16 + 8 * half]);
        }
        #pragma unroll
        for (int tj = 0; tj < 2; ++tj) {
            const int col = wn + tj * 16 + l15;
            bf[tj].q[0] = *reinterpret_cast<const uint4*>(&sB[col][kc + 8 * half]);
            bf[tj].q[1] = *reinterpret_cast<const uint4*>(&sB[col][kc + 16 + 8 * half]);
        }
        #pragma unroll
        for (int ti = 0; ti < 2; ++ti)
            #pragma unroll
            for (int tj = 0; tj < 2; ++tj)
                acc[ti][tj] = __builtin_amdgcn_wmma_f32_16x16x32_bf16(
                    false, a[ti].v, false, bf[tj].v, (short)0, acc[ti][tj], false, false);
    }

    #pragma unroll
    for (int ti = 0; ti < 2; ++ti)
        #pragma unroll
        for (int tj = 0; tj < 2; ++tj)
            #pragma unroll
            for (int j = 0; j < 8; ++j) {
                const int row = wm + ti * 16 + j + 8 * half;
                const int col = wn + tj * 16 + l15;
                Y[(size_t)(b * Nn + n0 + row) * DIM + h * DH + col] =
                    (__bf16)acc[ti][tj][j];
            }
}

// ---------------------------------------------------------------------------
// Host launcher
// ---------------------------------------------------------------------------
extern "C" void kernel_launch(void* const* d_in, const int* in_sizes, int n_in,
                              void* d_out, int out_size, void* d_ws, size_t ws_size,
                              hipStream_t stream) {
    const float* x      = (const float*)d_in[0];
    const float* gamma  = (const float*)d_in[1];
    const float* w_qkv  = (const float*)d_in[2];
    const float* temp   = (const float*)d_in[3];
    const float* w_out  = (const float*)d_in[4];

    float* out  = (float*)d_out;                       // [b, n, dim]
    float* vout = out + (size_t)Mtok * DIM;            // orig_v [b, h, d, n]

    char* ws = (char*)d_ws;
    size_t off = 0;
    auto take = [&](size_t bytes) -> char* {
        char* p = ws + off;
        off += (bytes + 255) & ~(size_t)255;
        return p;
    };
    __bf16* xn   = (__bf16*)take((size_t)Mtok * DIM * 2);       // 64 MB
    __bf16* wq   = (__bf16*)take((size_t)DIM * QKVC * 2);       // 6 MB
    __bf16* wo   = (__bf16*)take((size_t)DIM * DIM * 2);        // 2 MB
    float*  qkvf = (float*) take((size_t)Mtok * QKVC * 4);      // 384 MB
    float*  sumq = (float*) take((size_t)Bb * 1024 * 4);
    float*  sumk = (float*) take((size_t)Bb * 1024 * 4);
    float*  invq = (float*) take((size_t)Bb * 1024 * 4);
    float*  invk = (float*) take((size_t)Bb * 1024 * 4);
    __bf16* attn = (__bf16*)take((size_t)Bb * H * 64 * 64 * 2);
    __bf16* Yb   = (__bf16*)take((size_t)Mtok * DIM * 2);       // 64 MB

    // 1) RMSNorm -> bf16
    rmsnorm_bf16<<<dim3(Mtok), 256, 0, stream>>>(x, gamma, xn);
    // 2) weights -> bf16
    f32_to_bf16<<<dim3((DIM * QKVC) / 1024), 256, 0, stream>>>(w_qkv, wq, DIM * QKVC);
    f32_to_bf16<<<dim3((DIM * DIM) / 1024), 256, 0, stream>>>(w_out, wo, DIM * DIM);
    // 3) qkv = xn @ w_qkv   (M=32768, N=3072, K=1024), 128x128 tiles
    gemm_bf16<<<dim3(QKVC / 128, Mtok / 128), 256, 0, stream>>>(
        xn, DIM, wq, QKVC, qkvf, QKVC, DIM);
    // 4) channel norms of q,k over n
    zero_f32<<<dim3((2 * Bb * 1024) / 256), 256, 0, stream>>>(sumq, 2 * Bb * 1024);
    colnorm_partial<<<dim3(4, Bb, 8), 256, 0, stream>>>(qkvf, sumq, sumk);
    finalize_inv<<<dim3((Bb * 1024) / 256), 256, 0, stream>>>(sumq, sumk, invq, invk);
    // 5) orig_v output
    v_transpose<<<dim3(Nn / 64, (H * DH) / 64, Bb), 256, 0, stream>>>(qkvf, vout);
    // 6) sim + softmax -> attn (bf16)
    sim_softmax<<<dim3(H, Bb), 256, 0, stream>>>(qkvf, invq, invk, temp, attn);
    // 7) Y = blockdiag(attn) applied per token
    apply_attn<<<dim3(Nn / 128, H, Bb), 256, 0, stream>>>(qkvf, attn, Yb);
    // 8) out = Y @ w_out   (M=32768, N=1024, K=1024)
    gemm_bf16<<<dim3(DIM / 128, Mtok / 128), 256, 0, stream>>>(
        Yb, DIM, wo, DIM, out, DIM, DIM);

    (void)in_sizes; (void)n_in; (void)out_size; (void)ws_size;
}
